// Convolve_64888365908451
// MI455X (gfx1250) — matrix operand order repaired and checked by reading, not compiled, not run on hardware
//
#include <hip/hip_runtime.h>
#include <hip/hip_bf16.h>

// MI455X / gfx1250, wave32. Main contraction + epilogue run on
// V_WMMA_F32_16X16X32_BF16 with a 3-term bf16 hi/lo split (fp32-grade
// accuracy) so the kernel stays bound by the single 335 MB stream of A
// (~14.4 us at 23.3 TB/s) instead of the ~8x slower f32 WMMA path.
// This revision: double-buffered A-fragment LDS (1 barrier/iter, global
// loads for block jb+1 overlap WMMAs of block jb) + native packed bf16
// converts (v_cvt_pk_bf16_f32) instead of manual integer RNE.

typedef __attribute__((ext_vector_type(16))) __bf16 v16bf;
typedef __attribute__((ext_vector_type(2)))  __bf16 v2bf;
typedef __attribute__((ext_vector_type(8)))  float  v8f;

#define S_DIM  16
#define N_DIM  1024
#define R_DIM  4
#define C_IN   128
#define C_OUT  128
#define EPSF   1e-7f

// Split (f0,f1) into packed bf16 hi dword and residual lo dword.
// v2bf build from two f32 casts -> v_cvt_pk_bf16_f32.
__device__ __forceinline__ unsigned pack_hi_lo(float f0, float f1, unsigned &lo_out) {
    union { v2bf v; unsigned u; } H, L;
    __bf16 h0 = (__bf16)f0;
    __bf16 h1 = (__bf16)f1;
    H.v[0] = h0; H.v[1] = h1;
    L.v[0] = (__bf16)(f0 - (float)h0);
    L.v[1] = (__bf16)(f1 - (float)h1);
    lo_out = L.u;
    return H.u;
}

union VB { unsigned u[8]; v16bf v; };

__device__ __forceinline__ v8f wmma_bf16(v16bf a, v16bf b, v8f c) {
    // (neg_a, A, neg_b, B, c_mod, C, reuse_a, reuse_b)
    return __builtin_amdgcn_wmma_f32_16x16x32_bf16(false, a, false, b, (short)0, c, false, false);
}

// ---------------------------------------------------------------------------
// Pre-pass 1: pack x (S,N,128) f32 -> bf16 hi/lo, pairs of consecutive j per
// dword so a B-operand VGPR v at lane (h,n) is one coalesced dword load:
//   xp[(s*512 + j/2)*128 + a] = { bf16 x[j,a] , bf16 x[j+1,a] }
__global__ void pack_x_kernel(const float* __restrict__ x,
                              unsigned* __restrict__ xh, unsigned* __restrict__ xl) {
    int idx = blockIdx.x * 256 + threadIdx.x;          // [0, 16*512*128)
    int a  = idx & 127;
    int sj = idx >> 7;                                 // s*512 + jpair
    const float* p = x + ((size_t)sj * 2) * C_IN + a;
    unsigned lo;
    unsigned hi = pack_hi_lo(p[0], p[C_IN], lo);
    xh[idx] = hi; xl[idx] = lo;
}

// Pre-pass 2: build combined epilogue B matrix (640 x 128):
//   rows k<512: weight[a,b,c] with k = c*128 + a ; rows k>=512: theta[k-512, b]
// packed as k-pairs per dword (same layout rule as pack_x).
__global__ void pack_w_kernel(const float* __restrict__ w, const float* __restrict__ th,
                              unsigned* __restrict__ bh, unsigned* __restrict__ bl) {
    int idx = blockIdx.x * 256 + threadIdx.x;          // [0, 320*128)
    int b  = idx & 127;
    int kp = idx >> 7;                                 // 0..319
    int k  = kp * 2;
    float v0, v1;
    if (k < 512) {
        int c = k >> 7, a = k & 127;
        v0 = w[((size_t)a       * C_OUT + b) * R_DIM + c];
        v1 = w[((size_t)(a + 1) * C_OUT + b) * R_DIM + c];
    } else {
        v0 = th[(size_t)(k - 512) * C_OUT + b];
        v1 = th[(size_t)(k - 511) * C_OUT + b];
    }
    unsigned lo;
    unsigned hi = pack_hi_lo(v0, v1, lo);
    bh[idx] = hi; bl[idx] = lo;
}

// ---------------------------------------------------------------------------
// Main fused kernel: one WG per (s, 16-row i-tile); 8 waves, wave w owns
// a/b-columns [16w, 16w+16).
__global__ __launch_bounds__(256) void convolve_main(
    const float* __restrict__ A, const float* __restrict__ x,
    const unsigned* __restrict__ xph, const unsigned* __restrict__ xpl,
    const unsigned* __restrict__ bbh, const unsigned* __restrict__ bbl,
    float* __restrict__ out)
{
    // A fragments in exact WMMA A-operand layout: [buf][hi/lo][c][lane][word]
    __shared__ alignas(32) unsigned afrag[2][2][4][32][8];  // 16 KB, double-buffered
    __shared__ alignas(32) float    E[16][648];             // m (512) ++ x (128), padded
    __shared__ float                normLDS[16][4];

    const int tid = threadIdx.x;
    const int s  = blockIdx.x >> 6;
    const int i0 = (blockIdx.x & 63) << 4;

    // producer identity: row pm (0..15), j-pair pk (0..15) -> k = 2*pk, 2*pk+1
    const int pm = tid >> 4;
    const int pk = tid & 15;
    const int ph    = (pk >= 8) ? 1 : 0;                  // K-half k>=16
    const int plane = pm + 16 * ph;
    const int pwi   = pk - 8 * ph;

    // consumer identity
    const int lane = tid & 31;
    const int wave = tid >> 5;
    const int a0   = wave << 4;
    const int h    = lane >> 4;
    const int n    = lane & 15;

    if (tid < 64) normLDS[tid >> 2][tid & 3] = 0.0f;

    v8f   acc[4] = {};
    float ns[4]  = {0.f, 0.f, 0.f, 0.f};

    const float* aptr = A + (((size_t)s * N_DIM + (i0 + pm)) * N_DIM + 2 * pk) * 5;
    const unsigned* xhp = xph + ((size_t)s * 512 + 8 * h) * C_IN + a0 + n;
    const unsigned* xlp = xpl + ((size_t)s * 512 + 8 * h) * C_IN + a0 + n;

    // ---- pipeline prologue: stage block 0 into buffer 0 ----
    float ra[8];
    #pragma unroll
    for (int c = 0; c < 8; ++c) ra[c] = aptr[c < 4 ? c : c + 1];   // j and j+1, c=0..3
    aptr += 160;
    #pragma unroll
    for (int c = 0; c < 4; ++c) {
        ns[c] += ra[c] + ra[4 + c];
        unsigned lo;
        unsigned hi = pack_hi_lo(ra[c], ra[4 + c], lo);
        afrag[0][0][c][plane][pwi] = hi;
        afrag[0][1][c][plane][pwi] = lo;
    }
    __syncthreads();

    for (int jb = 0; jb < 32; ++jb) {
        const int cur = jb & 1;
        const bool have_next = (jb + 1 < 32);

        // -------- issue global loads for block jb+1 early (hide under WMMAs)
        float rn[8];
        if (have_next) {
            #pragma unroll
            for (int c = 0; c < 8; ++c) rn[c] = aptr[c < 4 ? c : c + 1];
            aptr += 160;
            __builtin_prefetch(aptr, 0, 0);
        }

        // -------- consumer: B frags from packed x (L2-hot), 12 WMMAs on buf cur
        VB bh, bl;
        #pragma unroll
        for (int v = 0; v < 8; ++v) { bh.u[v] = xhp[v * C_IN]; bl.u[v] = xlp[v * C_IN]; }
        xhp += 16 * C_IN; xlp += 16 * C_IN;

        #pragma unroll
        for (int c = 0; c < 4; ++c) {
            v16bf Ah = *reinterpret_cast<const v16bf*>(&afrag[cur][0][c][lane][0]);
            v16bf Al = *reinterpret_cast<const v16bf*>(&afrag[cur][1][c][lane][0]);
            acc[c] = wmma_bf16(Ah, bh.v, acc[c]);   // hi*hi
            acc[c] = wmma_bf16(Ah, bl.v, acc[c]);   // hi*lo
            acc[c] = wmma_bf16(Al, bh.v, acc[c]);   // lo*hi
        }

        // -------- producer: convert block jb+1, store into the other buffer
        if (have_next) {
            #pragma unroll
            for (int c = 0; c < 4; ++c) {
                ns[c] += rn[c] + rn[4 + c];
                unsigned lo;
                unsigned hi = pack_hi_lo(rn[c], rn[4 + c], lo);
                afrag[cur ^ 1][0][c][plane][pwi] = hi;
                afrag[cur ^ 1][1][c][plane][pwi] = lo;
            }
        }
        __syncthreads();
    }

    // -------- norm reduction (sum_j Ar) and raw m -> LDS
    #pragma unroll
    for (int c = 0; c < 4; ++c) atomicAdd(&normLDS[pm][c], ns[c]);

    #pragma unroll
    for (int r = 0; r < 8; ++r) {                 // C layout: lane row = r + 8h
        int i = r + 8 * h;
        E[i][0 * C_IN + a0 + n] = acc[0][r];
        E[i][1 * C_IN + a0 + n] = acc[1][r];
        E[i][2 * C_IN + a0 + n] = acc[2][r];
        E[i][3 * C_IN + a0 + n] = acc[3][r];
    }
    for (int t = tid; t < 16 * C_IN; t += 256) {  // theta input: E[:,512:640] = x
        int i = t >> 7, a = t & 127;
        E[i][512 + a] = x[((size_t)s * N_DIM + (i0 + i)) * C_IN + a];
    }
    __syncthreads();

    if (tid < 64) {
        float v = normLDS[tid >> 2][tid & 3];
        normLDS[tid >> 2][tid & 3] = 1.0f / (v + EPSF);
    }
    __syncthreads();

    for (int t = tid; t < 16 * 512; t += 256) {   // m *= norm (x part untouched)
        int i = t >> 9, k = t & 511;
        E[i][k] *= normLDS[i][k >> 7];
    }
    __syncthreads();

    // -------- epilogue matmul: (16 x 640) @ (640 x 128), 3-term bf16 split
    const int b0   = a0;
    const int mrow = n;
    v8f accO = {};
    for (int kc = 0; kc < 20; ++kc) {
        const int k0 = kc * 32;
        VB eh, el;
        #pragma unroll
        for (int ii = 0; ii < 4; ++ii) {          // words 0..3: K = k0+8h+2ii,+1
            float f0 = E[mrow][k0 + 8 * h + 2 * ii];
            float f1 = E[mrow][k0 + 8 * h + 2 * ii + 1];
            unsigned lo; unsigned hi = pack_hi_lo(f0, f1, lo);
            eh.u[ii] = hi; el.u[ii] = lo;
        }
        #pragma unroll
        for (int ii = 0; ii < 4; ++ii) {          // words 4..7: K = k0+16+8h+2ii,+1
            float f0 = E[mrow][k0 + 16 + 8 * h + 2 * ii];
            float f1 = E[mrow][k0 + 16 + 8 * h + 2 * ii + 1];
            unsigned lo; unsigned hi = pack_hi_lo(f0, f1, lo);
            eh.u[4 + ii] = hi; el.u[4 + ii] = lo;
        }
        VB wh, wl;
        const unsigned* bhp = bbh + ((size_t)(k0 >> 1) + 4 * h) * C_OUT + b0 + n;
        const unsigned* blp = bbl + ((size_t)(k0 >> 1) + 4 * h) * C_OUT + b0 + n;
        #pragma unroll
        for (int v = 0; v < 8; ++v) { wh.u[v] = bhp[(v + (v & 4)) * C_OUT]; wl.u[v] = blp[(v + (v & 4)) * C_OUT]; }
        accO = wmma_bf16(eh.v, wh.v, accO);
        accO = wmma_bf16(eh.v, wl.v, accO);
        accO = wmma_bf16(el.v, wh.v, accO);
    }

    #pragma unroll
    for (int r = 0; r < 8; ++r) {
        int i = r + 8 * h;
        out[((size_t)s * N_DIM + (i0 + i)) * C_OUT + b0 + n] = tanhf(accO[r]);
    }
}

// ---------------------------------------------------------------------------
extern "C" void kernel_launch(void* const* d_in, const int* in_sizes, int n_in,
                              void* d_out, int out_size, void* d_ws, size_t ws_size,
                              hipStream_t stream) {
    (void)in_sizes; (void)n_in; (void)out_size; (void)ws_size;
    const float* A     = (const float*)d_in[0];   // (16,1024,1024,5)
    const float* x     = (const float*)d_in[1];   // (16,1024,128)
    const float* w     = (const float*)d_in[2];   // (128,128,4)
    const float* theta = (const float*)d_in[3];   // (128,128)
    float* out = (float*)d_out;                   // (16,1024,128)

    // workspace carve: 2 x 4 MB packed x + 2 x 160 KB packed (weight||theta)
    unsigned* xph = (unsigned*)d_ws;
    unsigned* xpl = xph + (size_t)S_DIM * 512 * C_IN;
    unsigned* bbh = xpl + (size_t)S_DIM * 512 * C_IN;
    unsigned* bbl = bbh + (size_t)320 * C_OUT;

    pack_x_kernel<<<(S_DIM * 512 * C_IN) / 256, 256, 0, stream>>>(x, xph, xpl);
    pack_w_kernel<<<(320 * C_OUT) / 256, 256, 0, stream>>>(w, theta, bbh, bbl);
    convolve_main<<<S_DIM * (N_DIM / 16), 256, 0, stream>>>(A, x, xph, xpl, bbh, bbl, out);
}
